// SelfMultiheadAttn_72129680769539
// MI455X (gfx1250) — compile-verified, matrix operand order checked by reading, not run on hardware
//
#include <hip/hip_runtime.h>
#include <hip/hip_bf16.h>
#include <stdint.h>

// ---------------- problem constants ----------------
#define T_SEQ 2048
#define BATCH 2
#define EMB   1024
#define NH    16
#define HD    64
#define M_ROWS (T_SEQ * BATCH)   // 4096 rows for both GEMMs
#define N_QKV  (3 * EMB)         // 3072
#define BIG_NEG (-1.0e30f)
#define SSTR 2052                // LDS score row stride (floats), padded vs 2048

typedef __attribute__((ext_vector_type(16))) __bf16 bf16x16;
typedef __attribute__((ext_vector_type(8)))  float  v8f;
typedef int v4i __attribute__((vector_size(16)));   // matches builtin param type

#define AS1 __attribute__((address_space(1)))
#define AS3 __attribute__((address_space(3)))

#if defined(__has_builtin)
#  if __has_builtin(__builtin_amdgcn_global_load_async_to_lds_b128)
#    define HAVE_ASYNC_LDS 1
#  else
#    define HAVE_ASYNC_LDS 0
#  endif
#  if __has_builtin(__builtin_amdgcn_s_wait_asynccnt)
#    define HAVE_WAIT_ASYNC 1
#  else
#    define HAVE_WAIT_ASYNC 0
#  endif
#else
#  define HAVE_ASYNC_LDS 0
#  define HAVE_WAIT_ASYNC 0
#endif

union FragBF {
  bf16x16 v;
  unsigned short u[16];
  uint4 q[2];
};

__device__ __forceinline__ unsigned short f2bf(float f) {
  unsigned int u = __builtin_bit_cast(unsigned int, f);
  return (unsigned short)((u + 0x7FFFu + ((u >> 16) & 1u)) >> 16);  // RNE
}
__device__ __forceinline__ float bf2f(unsigned short h) {
  unsigned int u = ((unsigned int)h) << 16;
  return __builtin_bit_cast(float, u);
}
// convert 8 consecutive fp32 (32B) -> 8 bf16
__device__ __forceinline__ void cvt8(const float4* p, unsigned short* d) {
  float4 a = p[0], b = p[1];
  d[0] = f2bf(a.x); d[1] = f2bf(a.y); d[2] = f2bf(a.z); d[3] = f2bf(a.w);
  d[4] = f2bf(b.x); d[5] = f2bf(b.y); d[6] = f2bf(b.z); d[7] = f2bf(b.w);
}
// A/B 16-bit fragment: elems 0..7 <- base..base+7, elems 8..15 <- base+16..base+23
__device__ __forceinline__ bf16x16 load_bf_frag(const unsigned short* c0,
                                                const unsigned short* c1) {
  FragBF f;
  f.q[0] = *(const uint4*)c0;
  f.q[1] = *(const uint4*)c1;
  return f.v;
}
__device__ __forceinline__ v8f wmma_bf16(bf16x16 a, bf16x16 b, v8f c) {
  return __builtin_amdgcn_wmma_f32_16x16x32_bf16(
      /*neg_a=*/false, a, /*neg_b=*/false, b,
      /*c_mod=*/(short)0, c, /*reuse_a=*/false, /*reuse_b=*/false);
}
__device__ __forceinline__ void async_wait_all() {
#if HAVE_ASYNC_LDS
#  if HAVE_WAIT_ASYNC
  __builtin_amdgcn_s_wait_asynccnt(0);
#  else
  asm volatile("s_wait_asynccnt 0x0" ::: "memory");
#  endif
#endif
}

// ---------------- kernel 1: fp32 -> bf16 conversion ----------------
__global__ void f32_to_bf16_kernel(const float* __restrict__ src,
                                   unsigned short* __restrict__ dst, int n) {
  int i = blockIdx.x * blockDim.x + threadIdx.x;
  if (i < n) dst[i] = f2bf(src[i]);
}

// ---------------- kernel 2/6: GEMM  out[M,N] = A[M,K] * W[N,K]^T + bias ----
// A bf16 [M,K], W bf16 [N,K]. WG = 8 waves: M-tile 128 (16 per wave), N-tile 64.
// W fragments for the 4 n-subtiles are staged into LDS once per K-step (async,
// double-buffered) and shared by all 8 waves.
template <bool STORE_BF16>
__global__ void gemm_wmma_kernel(const unsigned short* __restrict__ A,
                                 const unsigned short* __restrict__ W,
                                 const float* __restrict__ bias,
                                 void* __restrict__ out, int M, int N, int K) {
  // staging: [buf][ntile][lane*16 bf16] ; fragment-ready layout, 8 KB total
  __shared__ __align__(16) unsigned short wstage[2][4][32 * 16];

  const int lane = threadIdx.x & 31;
  const int wid  = threadIdx.x >> 5;
  const int r    = lane & 15;
  const int half = lane >> 4;
  const int m0   = blockIdx.y * 128 + wid * 16;
  const int n0   = blockIdx.x * 64;

  v8f acc[4];
  #pragma unroll
  for (int nt = 0; nt < 4; nt++)
    acc[nt] = v8f{0.f, 0.f, 0.f, 0.f, 0.f, 0.f, 0.f, 0.f};

  const unsigned short* arow = A + (size_t)(m0 + r) * K;

  // wave 0 stages W fragments for K-step starting at k0 into buffer bufi
  auto stage = [&](int k0, int bufi) {
    if (wid == 0) {
      #pragma unroll
      for (int nt = 0; nt < 4; nt++) {
        const unsigned short* wp =
            W + (size_t)(n0 + nt * 16 + r) * K + k0 + half * 8;
        unsigned short* lb = &wstage[bufi][nt][lane * 16];
#if HAVE_ASYNC_LDS
        __builtin_amdgcn_global_load_async_to_lds_b128(
            (AS1 v4i*)wp, (AS3 v4i*)lb, 0, 0);
        __builtin_amdgcn_global_load_async_to_lds_b128(
            (AS1 v4i*)(wp + 16), (AS3 v4i*)(lb + 8), 0, 0);
#else
        *(uint4*)lb       = *(const uint4*)wp;
        *(uint4*)(lb + 8) = *(const uint4*)(wp + 16);
#endif
      }
    }
  };

  const int nsteps = K / 32;
  stage(0, 0);
  async_wait_all();
  __syncthreads();

  for (int ks = 0; ks < nsteps; ks++) {
    const int k0  = ks * 32;
    const int buf = ks & 1;
    if (ks + 1 < nsteps) stage(k0 + 32, buf ^ 1);  // overlap next stage

    const int ca = k0 + half * 8;
    bf16x16 af = load_bf_frag(arow + ca, arow + ca + 16);
    __builtin_prefetch(arow + ca + 64, 0, 0);  // -> global_prefetch_b8

    #pragma unroll
    for (int nt = 0; nt < 4; nt++) {
      const unsigned short* lb = &wstage[buf][nt][lane * 16];
      FragBF bf;
      bf.q[0] = *(const uint4*)lb;        // ds_load_b128
      bf.q[1] = *(const uint4*)(lb + 8);  // ds_load_b128
      acc[nt] = wmma_bf16(af, bf.v, acc[nt]);
    }
    async_wait_all();
    __syncthreads();  // staged buffer ready; readers done with old buffer
  }

  #pragma unroll
  for (int nt = 0; nt < 4; nt++) {
    #pragma unroll
    for (int e = 0; e < 8; e++) {
      const int m = m0 + e + 8 * half;  // C layout: M = e (+8 for hi half)
      const int n = n0 + nt * 16 + r;   // N = lane
      const float v = acc[nt][e] + bias[n];
      if (STORE_BF16)
        ((unsigned short*)out)[(size_t)m * N + n] = f2bf(v);
      else
        ((float*)out)[(size_t)m * N + n] = v;
    }
  }
}

// ---------------- kernel 3: RoPE + head-layout + V transpose ----------------
// qkv bf16 [T,B,3E] -> Q,K bf16 [B,H,T,D] (Q pre-scaled by D^-0.5), Vt bf16 [B,H,D,T]
__global__ void rope_kernel(const unsigned short* __restrict__ qkv,
                            const float* __restrict__ pos,
                            unsigned short* __restrict__ Qb,
                            unsigned short* __restrict__ Kb,
                            unsigned short* __restrict__ Vt) {
  const int idx = blockIdx.x * blockDim.x + threadIdx.x;
  const int d  = idx & (HD - 1);
  const int t  = (idx >> 6) & (T_SEQ - 1);
  const int hh = (idx >> 17) & (NH - 1);
  const int b  = idx >> 21;
  const size_t row = ((size_t)t * BATCH + b) * N_QKV;
  const int col = hh * HD + d;
  const int bh = b * NH + hh;
  // V: transpose copy (bf16 passthrough)
  Vt[((size_t)bh * HD + d) * T_SEQ + t] = qkv[row + 2 * EMB + col];
  if (d < 32) {
    const float ang = pos[t * HD + d];           // pos[:,d]==pos[:,d+32]
    const float cs = cosf(ang), sn = sinf(ang);
    const float q1 = bf2f(qkv[row + col]);
    const float q2 = bf2f(qkv[row + col + 32]);
    const float k1 = bf2f(qkv[row + EMB + col]);
    const float k2 = bf2f(qkv[row + EMB + col + 32]);
    const float scale = 0.125f;                  // 1/sqrt(64) folded into Q
    const size_t qi = ((size_t)bh * T_SEQ + t) * HD + d;
    Qb[qi]      = f2bf((q1 * cs - q2 * sn) * scale);
    Qb[qi + 32] = f2bf((q2 * cs + q1 * sn) * scale);
    Kb[qi]      = f2bf(k1 * cs - k2 * sn);
    Kb[qi + 32] = f2bf(k2 * cs + k1 * sn);
  }
}

// ---------------- kernel 4: fused scores + softmax + attn-out + P*V ---------
// One WG (8 waves) per (b,h, 16-row tile). LDS holds 16x2048 fp32 score strip.
__global__ void attn_ctx_kernel(const unsigned short* __restrict__ Qb,
                                const unsigned short* __restrict__ Kb,
                                const unsigned short* __restrict__ Vt,
                                float* __restrict__ attn,
                                float* __restrict__ ctx) {
  extern __shared__ float smem[];
  float* sc    = smem;               // 16 * SSTR
  float* red   = sc + 16 * SSTR;     // 16 * 16 (max/sum partials)
  float* cpart = red + 256;          // 4 tiles * 32 lanes * 8 = 1024

  const int lane = threadIdx.x & 31;
  const int wid  = threadIdx.x >> 5;
  const int r    = lane & 15;
  const int half = lane >> 4;
  const int t0   = blockIdx.x * 16;
  const int bh   = blockIdx.y;       // b*NH + h
  const int b    = bh >> 4;
  const int h    = bh & 15;

  // ---- Q fragments (16 rows x 64, two K=32 steps), reused across 16 col tiles
  const unsigned short* qrow = Qb + ((size_t)bh * T_SEQ + t0 + r) * HD;
  const bf16x16 aq0 = load_bf_frag(qrow + half * 8,      qrow + 16 + half * 8);
  const bf16x16 aq1 = load_bf_frag(qrow + 32 + half * 8, qrow + 48 + half * 8);

  // ---- phase 1: scores = Q*K^T into LDS, software-pipelined over col tiles.
  // Tiles this wave owns: ct = wid + 8*i, i=0..15; active (c0 <= t0) is a prefix.
  const int tmax = t0 >> 4;
  const int nact = (wid <= tmax) ? ((tmax - wid) >> 3) + 1 : 0;

  // fill the fully-masked region first
  for (int i = nact; i < 16; i++) {
    const int c0 = (wid + 8 * i) * 16;
    #pragma unroll
    for (int e = 0; e < 8; e++)
      sc[(e + 8 * half) * SSTR + c0 + r] = BIG_NEG;
  }
  if (nact > 0) {
    const unsigned short* kbase = Kb + (size_t)bh * T_SEQ * HD;
    // prologue: load tile 0 fragments
    {
      const int c0 = wid * 16;
      const unsigned short* krow = kbase + (size_t)(c0 + r) * HD;
      const bf16x16 f0 = load_bf_frag(krow + half * 8,      krow + 16 + half * 8);
      const bf16x16 f1 = load_bf_frag(krow + 32 + half * 8, krow + 48 + half * 8);
      bf16x16 b0 = f0, b1 = f1;
      for (int i = 0; i < nact; i++) {
        const bf16x16 cb0 = b0, cb1 = b1;
        const int c0i = (wid + 8 * i) * 16;
        if (i + 1 < nact) {          // issue next tile's loads before consuming
          const int c1 = (wid + 8 * (i + 1)) * 16;
          const unsigned short* krow2 = kbase + (size_t)(c1 + r) * HD;
          b0 = load_bf_frag(krow2 + half * 8,      krow2 + 16 + half * 8);
          b1 = load_bf_frag(krow2 + 32 + half * 8, krow2 + 48 + half * 8);
        }
        v8f s = v8f{0.f, 0.f, 0.f, 0.f, 0.f, 0.f, 0.f, 0.f};
        s = wmma_bf16(aq0, cb0, s);
        s = wmma_bf16(aq1, cb1, s);
        if (c0i == t0) {             // diagonal tile: per-element causal mask
          #pragma unroll
          for (int e = 0; e < 8; e++) {
            const int m = e + 8 * half;
            sc[m * SSTR + c0i + r] = (r > m) ? BIG_NEG : s[e];
          }
        } else {
          #pragma unroll
          for (int e = 0; e < 8; e++)
            sc[(e + 8 * half) * SSTR + c0i + r] = s[e];
        }
      }
    }
  }
  __syncthreads();

  // ---- phase 2: softmax over each of 16 rows (16 threads/row, stride 16)
  {
    const int row = threadIdx.x >> 4;
    const int j   = threadIdx.x & 15;
    float* srow = sc + row * SSTR;
    float mx = BIG_NEG;
    for (int i = 0; i < 128; i++) mx = fmaxf(mx, srow[j + 16 * i]);
    red[row * 16 + j] = mx;
    __syncthreads();
    float rm = red[row * 16];
    #pragma unroll
    for (int p = 1; p < 16; p++) rm = fmaxf(rm, red[row * 16 + p]);
    float sum = 0.f;
    for (int i = 0; i < 128; i++) {
      const float e = __expf(srow[j + 16 * i] - rm);
      srow[j + 16 * i] = e;
      sum += e;
    }
    __syncthreads();
    red[row * 16 + j] = sum;
    __syncthreads();
    float rs = 0.f;
    #pragma unroll
    for (int p = 0; p < 16; p++) rs += red[row * 16 + p];
    const float inv = 1.0f / rs;
    float* ag = attn + ((size_t)bh * T_SEQ + t0 + row) * T_SEQ;
    for (int i = 0; i < 128; i++) {
      const float p = srow[j + 16 * i] * inv;
      srow[j + 16 * i] = p;          // keep P in LDS for phase 3
      ag[j + 16 * i] = p;            // dominant HBM write (attn output)
    }
  }
  __syncthreads();

  // ---- phase 3: ctx = P[16x2048] * V[2048x64]; 8 waves = 4 N-tiles x 2 K-halves
  // Software-pipelined over K-steps with causal bound folded into loop count.
  {
    const int ntile = wid & 3;
    const int khalf = wid >> 2;
    const int d0 = ntile * 16;
    const unsigned short* vrow = Vt + ((size_t)bh * HD + d0 + r) * T_SEQ;
    const float* prow = sc + r * SSTR;
    const int klast = (t0 + 15) >> 5;              // last K-step with P != 0
    const int kbeg  = khalf * 32;
    const int kend  = min(khalf * 32 + 32, klast + 1);
    v8f c = v8f{0.f, 0.f, 0.f, 0.f, 0.f, 0.f, 0.f, 0.f};
    if (kend > kbeg) {
      int ca = kbeg * 32 + half * 8;
      bf16x16 bv = load_bf_frag(vrow + ca, vrow + ca + 16);
      FragBF pa;
      cvt8((const float4*)(prow + ca), pa.u);
      cvt8((const float4*)(prow + ca + 16), pa.u + 8);
      for (int ks = kbeg; ks < kend; ks++) {
        const bf16x16 cbv = bv;
        const FragBF cpa = pa;
        if (ks + 1 < kend) {         // issue next V load + P cvt before WMMA
          const int cn = (ks + 1) * 32 + half * 8;
          bv = load_bf_frag(vrow + cn, vrow + cn + 16);
          cvt8((const float4*)(prow + cn), pa.u);
          cvt8((const float4*)(prow + cn + 16), pa.u + 8);
        }
        c = wmma_bf16(cpa.v, cbv, c);
      }
    }
    if (khalf == 1) {
      #pragma unroll
      for (int e = 0; e < 8; e++) cpart[ntile * 256 + lane * 8 + e] = c[e];
    }
    __syncthreads();
    if (khalf == 0) {
      #pragma unroll
      for (int e = 0; e < 8; e++) {
        const float v = c[e] + cpart[ntile * 256 + lane * 8 + e];
        const int m = t0 + e + 8 * half;
        ctx[((size_t)m * BATCH + b) * EMB + h * HD + d0 + r] = v;
      }
    }
  }
}

// ---------------- host-side launch ----------------
extern "C" void kernel_launch(void* const* d_in, const int* in_sizes, int n_in,
                              void* d_out, int out_size, void* d_ws, size_t ws_size,
                              hipStream_t stream) {
  const float* x    = (const float*)d_in[0];  // [T,B,E]
  const float* pos  = (const float*)d_in[1];  // [T,D]
  // d_in[2]: attn_mask (causal; handled analytically)
  const float* wqkv = (const float*)d_in[3];  // [3E,E]
  const float* bqkv = (const float*)d_in[4];  // [3E]
  const float* wout = (const float*)d_in[5];  // [E,E]
  const float* bout = (const float*)d_in[6];  // [E]

  char* ws = (char*)d_ws;
  unsigned short* wq_bf  = (unsigned short*)ws; ws += (size_t)N_QKV * EMB * 2;
  unsigned short* wo_bf  = (unsigned short*)ws; ws += (size_t)EMB * EMB * 2;
  unsigned short* x_bf   = (unsigned short*)ws; ws += (size_t)M_ROWS * EMB * 2;
  unsigned short* qkv_bf = (unsigned short*)ws; ws += (size_t)M_ROWS * N_QKV * 2;
  unsigned short* Qb     = (unsigned short*)ws; ws += (size_t)BATCH * NH * T_SEQ * HD * 2;
  unsigned short* Kb     = (unsigned short*)ws; ws += (size_t)BATCH * NH * T_SEQ * HD * 2;
  unsigned short* Vt     = (unsigned short*)ws; ws += (size_t)BATCH * NH * T_SEQ * HD * 2;
  float*          ctx    = (float*)ws;          ws += (size_t)M_ROWS * EMB * 4;
  unsigned short* ctx_bf = (unsigned short*)ws; ws += (size_t)M_ROWS * EMB * 2;

  float* out_main = (float*)d_out;                               // [T,B,E]
  float* attn_out = (float*)d_out + (size_t)T_SEQ * BATCH * EMB; // [B,H,T,T]

  f32_to_bf16_kernel<<<(N_QKV * EMB) / 256, 256, 0, stream>>>(wqkv, wq_bf, N_QKV * EMB);
  f32_to_bf16_kernel<<<(EMB * EMB) / 256, 256, 0, stream>>>(wout, wo_bf, EMB * EMB);
  f32_to_bf16_kernel<<<(M_ROWS * EMB) / 256, 256, 0, stream>>>(x, x_bf, M_ROWS * EMB);

  gemm_wmma_kernel<true><<<dim3(N_QKV / 64, M_ROWS / 128), 256, 0, stream>>>(
      x_bf, wq_bf, bqkv, qkv_bf, M_ROWS, N_QKV, EMB);

  rope_kernel<<<(BATCH * NH * T_SEQ * HD) / 256, 256, 0, stream>>>(
      qkv_bf, pos, Qb, Kb, Vt);

  const size_t smem = (size_t)(16 * SSTR + 256 + 1024) * sizeof(float); // ~136 KB
  attn_ctx_kernel<<<dim3(T_SEQ / 16, BATCH * NH), 256, smem, stream>>>(
      Qb, Kb, Vt, attn_out, ctx);

  f32_to_bf16_kernel<<<(M_ROWS * EMB) / 256, 256, 0, stream>>>(ctx, ctx_bf, M_ROWS * EMB);

  gemm_wmma_kernel<false><<<dim3(EMB / 64, M_ROWS / 128), 256, 0, stream>>>(
      ctx_bf, wo_bf, bout, out_main, M_ROWS, EMB, EMB);
}
